// Imager_7473243095684
// MI455X (gfx1250) — matrix operand order repaired and checked by reading, not compiled, not run on hardware
//
#include <hip/hip_runtime.h>

typedef __attribute__((ext_vector_type(16))) _Float16 v16h;
typedef __attribute__((ext_vector_type(8)))  _Float16 v8h;
typedef __attribute__((ext_vector_type(8)))  float    v8f;

#define NPART  50000
#define NB     128
#define BATCH  8
#define KC     128       // particle chunk staged per produce/consume cycle
#define KSLICE 2048      // K range per block (16 chunks)
#define NSPLIT 25        // ceil(NPART / KSLICE) -> 8*25 = 200 blocks

#define CAT16 0,1,2,3,4,5,6,7,8,9,10,11,12,13,14,15

// Raw v_exp_f32 (no OCML ldexp/range guard; our arg is always <= 0 and
// anything below -126 is sub-f16-precision -> 0 is the desired result).
#if __has_builtin(__builtin_amdgcn_exp2f)
#define FAST_EXP2(x) __builtin_amdgcn_exp2f(x)
#else
#define FAST_EXP2(x) exp2f(x)
#endif

// ---------------------------------------------------------------------------
// Zero output + per-batch sums (harness poisons them; we accumulate atomically).
// ---------------------------------------------------------------------------
__global__ __launch_bounds__(256)
void kde_zero_kernel(float* __restrict__ out, float* __restrict__ sums) {
    const int i = blockIdx.x * blockDim.x + threadIdx.x;
    if (i < BATCH * NB * NB) out[i] = 0.0f;
    if (i < BATCH)           sums[i] = 0.0f;
}

// ---------------------------------------------------------------------------
// Fused KDE + joint-histogram GEMM, produce-once / consume-8x via LDS.
// Grid: (batch=8, kslice=25); block = 256 threads = 8 wave32s.
//
// Producer: thread t owns (axis = t>>7, bin = t&127); computes
//   kv[axis][bin][k] = exp2((x[axis][n0+k]-bins[bin])^2 * C), f16,
//   C = -0.5*log2(e)/h^2  (folded: v_sub, v_mul, v_mul, v_exp_f32)
// stored in LDS with K contiguous per bin -> fragment loads are ds_load_b128.
//
// Consumer: wave w owns c_tile = w, all 8 d_tiles:
//   acc[dt] += A(c_tile) x B(dt)  via v_wmma_f32_16x16x32_f16
// A frag (16x32 f16): lane m=lane&15, VGPR v -> K = (v&3)*2 + half*8 + (v>>2)*16
//   == two contiguous 8-f16 runs at k = ks+half*8 and k = ks+half*8+16.
// B frag (32x16 f16): lane n=lane&15, VGPR v -> K = (v&3)*2 + half*8 + (v>>2)*16
//   == two contiguous 8-f16 runs at k = ks+half*8 and k = ks+half*8+16 (as A).
// ---------------------------------------------------------------------------
__global__ __launch_bounds__(256)
void kde_joint_kernel(const float* __restrict__ X,
                      const float* __restrict__ bins,
                      const float* __restrict__ bwptr,
                      float* __restrict__ out,
                      float* __restrict__ batch_sums)
{
    __shared__ float sx[2][KC];                          // raw particle chunk (1 KB)
    __shared__ __align__(16) _Float16 skv[2][NB][KC];    // 64 KB kernel values

    const int b      = blockIdx.x;
    const int kbase  = blockIdx.y * KSLICE;
    const int t      = threadIdx.x;
    const int wave   = t >> 5;          // c_tile
    const int lane   = t & 31;
    const int halfid = lane >> 4;
    const int m      = lane & 15;

    const float bw   = bwptr[0];
    // exp(-0.5*((x-b)/h)^2) == exp2((x-b)^2 * cexp)
    const float cexp = -0.72134752044448170f / (bw * bw);  // -0.5*log2(e)/h^2

    // producer identity
    const int   paxis = t >> 7;         // 0: kv0(x0), 1: kv1(x1)
    const int   pbin  = t & 127;
    const float pbinv = bins[pbin];

    v8f acc[8];
#pragma unroll
    for (int dt = 0; dt < 8; ++dt) acc[dt] = (v8f){};

    for (int c = 0; c < KSLICE / KC; ++c) {
        const int n0 = kbase + c * KC;

        // ---- stage raw X chunk: one value per thread (sentinel pads tail) ----
        {
            const int a   = t >> 7;            // axis
            const int i   = t & (KC - 1);
            const int idx = n0 + i;
            const float* xs = X + (size_t)(a * BATCH + b) * NPART;
            sx[a][i] = (idx < NPART) ? xs[idx] : 1e18f;   // (1e18)^2*cexp -> -inf -> 0
        }
        __syncthreads();

        // ---- produce: KC exps per thread, vectorized f16 stores ----
#pragma unroll
        for (int k8 = 0; k8 < KC; k8 += 8) {
            v8h pk;
#pragma unroll
            for (int j = 0; j < 8; ++j) {
                const float d = sx[paxis][k8 + j] - pbinv;
                pk[j] = (_Float16)FAST_EXP2(d * d * cexp);
            }
            *(v8h*)&skv[paxis][pbin][k8] = pk;           // ds_store_b128
        }
        __syncthreads();

        // ---- consume: 4 k-steps x 8 d_tiles = 32 WMMAs per chunk ----
#pragma unroll
        for (int ks = 0; ks < KC; ks += 32) {
            const _Float16* arow = &skv[0][wave * 16 + m][ks + halfid * 8];
            const v8h alo = *(const v8h*)(arow);          // K = ks+half*8 .. +7
            const v8h ahi = *(const v8h*)(arow + 16);     // K = +16 .. +23
            const v16h afrag = __builtin_shufflevector(alo, ahi, CAT16);
#pragma unroll
            for (int dt = 0; dt < 8; ++dt) {
                const _Float16* brow = &skv[1][dt * 16 + m][ks + halfid * 8];
                const v8h blo = *(const v8h*)(brow);      // K = ks+half*8 .. +7
                const v8h bhi = *(const v8h*)(brow + 16); // K = +16 .. +23
                const v16h bfrag = __builtin_shufflevector(blo, bhi, CAT16);
                acc[dt] = __builtin_amdgcn_wmma_f32_16x16x32_f16(
                    false, afrag, false, bfrag, (short)0, acc[dt], false, false);
            }
        }
        __syncthreads();
    }

    // ---- accumulate partial tiles + batch sum ----
    // C/D layout: VGPR r -> row M = r + 8*half, col N = lane%16
    float tsum = 0.0f;
#pragma unroll
    for (int dt = 0; dt < 8; ++dt) {
#pragma unroll
        for (int r = 0; r < 8; ++r) {
            const int row = wave * 16 + r + 8 * halfid;
            const int col = dt * 16 + m;
            atomicAdd(&out[((size_t)b * NB + row) * NB + col], acc[dt][r]);
            tsum += acc[dt][r];
        }
    }
#pragma unroll
    for (int off = 16; off >= 1; off >>= 1)
        tsum += __shfl_xor(tsum, off, 32);
    if (lane == 0)
        atomicAdd(&batch_sums[b], tsum);
}

// ---------------------------------------------------------------------------
// Normalize: out[b,c,d] /= (sum[b] + EPS)
// ---------------------------------------------------------------------------
__global__ __launch_bounds__(256)
void kde_norm_kernel(float* __restrict__ out, const float* __restrict__ sums) {
    const int i = blockIdx.x * blockDim.x + threadIdx.x;
    if (i < BATCH * NB * NB) {
        const int b = i / (NB * NB);
        out[i] = out[i] / (sums[b] + 1e-10f);
    }
}

extern "C" void kernel_launch(void* const* d_in, const int* in_sizes, int n_in,
                              void* d_out, int out_size, void* d_ws, size_t ws_size,
                              hipStream_t stream) {
    const float* X    = (const float*)d_in[0];   // (2, 8, 50000) f32
    const float* bins = (const float*)d_in[1];   // (128,) f32
    const float* bw   = (const float*)d_in[2];   // scalar f32
    float* out  = (float*)d_out;                 // (8, 128, 128) f32
    float* sums = (float*)d_ws;                  // 8 floats of scratch

    const int total = BATCH * NB * NB;           // 131072

    kde_zero_kernel<<<(total + 255) / 256, 256, 0, stream>>>(out, sums);

    dim3 grid(BATCH, NSPLIT);                    // (8, 25) = 200 blocks
    kde_joint_kernel<<<grid, 256, 0, stream>>>(X, bins, bw, out, sums);

    kde_norm_kernel<<<(total + 255) / 256, 256, 0, stream>>>(out, sums);
}
